// Decoder_74955769250283
// MI455X (gfx1250) — matrix operand order repaired and checked by reading, not compiled, not run on hardware
//
#include <hip/hip_runtime.h>

#define BB 32
#define TT 32
#define HH 512
#define EE 512
#define KK 512
#define VDIM 512
#define BOOMD 2048
#define NWORD 32000
#define NATT 64
#define DATT 1024
#define NSTEP 31
#define NROWS (NSTEP*BB)      /* 992 */
#define NSTRIPE 64
#define NWG (NWORD/NSTRIPE)   /* 500 */

typedef float    v2f   __attribute__((ext_vector_type(2)));
typedef float    v8f   __attribute__((ext_vector_type(8)));
typedef __bf16   v16bf __attribute__((ext_vector_type(16)));
typedef unsigned uvec4 __attribute__((ext_vector_type(4)));

__device__ __forceinline__ float sigm(float x){ return 1.0f/(1.0f+__expf(-x)); }
__device__ __forceinline__ float fast_tanh(float x){
  float t = __expf(2.0f*x);
  return (t - 1.0f) / (t + 1.0f);
}
__device__ __forceinline__ unsigned short f2bf(float x){
  unsigned u = __float_as_uint(x);
  u += 0x7fffu + ((u>>16)&1u);          // round-to-nearest-even bf16
  return (unsigned short)(u>>16);
}

// ---------------------------------------------------------------------------
// Embedding gather + label extraction.  row = t*32 + b  (t in [0,31))
// ---------------------------------------------------------------------------
__global__ __launch_bounds__(256) void k_gather(const int* __restrict__ y,
    const float* __restrict__ embW, float* __restrict__ x_all,
    int* __restrict__ labels){
  int row = blockIdx.x;               // 0..991
  int t = row >> 5, b = row & 31;
  int tok = y[b*TT + t];
  if (threadIdx.x == 0) labels[row] = y[b*TT + t + 1];
  for (int j = threadIdx.x; j < EE; j += 256)
    x_all[(size_t)row*EE + j] = embW[(size_t)tok*EE + j];
}

// ---------------------------------------------------------------------------
// Generic f32 GEMM via V_WMMA_F32_16X16X4_F32, one wave per 16x16 output tile.
// C = A*B (+ A2*B2) (+ bias + bias2), optional silu-gate activation.
// Inner loop uses strided pointer increments: A pair -> one global_load_b64,
// B rows via two incremented base pointers (no per-iteration 64-bit muls).
// A-frag (16x4): vgpr0 = K(2*half), vgpr1 = K(2*half+1), lane m = row.
// B-frag (4x16): same K split per lane-half, n = lane&15.
// C/D: vgpr r -> rows r (lanes 0-15) / r+8 (lanes 16-31), col = lane&15.
// ---------------------------------------------------------------------------
__global__ __launch_bounds__(32) void k_gemm(const float* __restrict__ A, int lda,
    const float* __restrict__ Bm, int ldb,
    const float* __restrict__ A2, int lda2,
    const float* __restrict__ B2, int ldb2,
    const float* __restrict__ bias, const float* __restrict__ bias2,
    float* __restrict__ C, int ldc, int Kd, int Kd2, int act){
  const int lane = threadIdx.x & 31;
  const int m    = lane & 15;
  const int half = lane >> 4;
  const int arow = blockIdx.y*16 + m;
  const int bcol = blockIdx.x*16 + m;
  float accf[8];
#if __has_builtin(__builtin_amdgcn_wmma_f32_16x16x4_f32)
  v8f acc = {};
  {
    const float* ap  = A  + (size_t)arow*lda + 2*half;
    const float* bp0 = Bm + (size_t)(2*half)*ldb + bcol;
    const float* bp1 = bp0 + ldb;
    const size_t bstep = 4*(size_t)ldb;
    #pragma unroll 4
    for (int k0 = 0; k0 < Kd; k0 += 4){
      v2f a = *(const v2f*)ap;
      v2f b; b.x = *bp0; b.y = *bp1;
      acc = __builtin_amdgcn_wmma_f32_16x16x4_f32(false, a, false, b,
                                                  (short)0, acc, false, false);
      ap += 4; bp0 += bstep; bp1 += bstep;
    }
  }
  if (A2){
    const float* ap  = A2 + (size_t)arow*lda2 + 2*half;
    const float* bp0 = B2 + (size_t)(2*half)*ldb2 + bcol;
    const float* bp1 = bp0 + ldb2;
    const size_t bstep = 4*(size_t)ldb2;
    #pragma unroll 4
    for (int k0 = 0; k0 < Kd2; k0 += 4){
      v2f a = *(const v2f*)ap;
      v2f b; b.x = *bp0; b.y = *bp1;
      acc = __builtin_amdgcn_wmma_f32_16x16x4_f32(false, a, false, b,
                                                  (short)0, acc, false, false);
      ap += 4; bp0 += bstep; bp1 += bstep;
    }
  }
  #pragma unroll
  for (int r = 0; r < 8; ++r) accf[r] = acc[r];
#else
  for (int r = 0; r < 8; ++r){
    int orow = blockIdx.y*16 + r + 8*half;
    float s = 0.f;
    for (int k = 0; k < Kd; ++k)
      s += A[(size_t)orow*lda + k] * Bm[(size_t)k*ldb + bcol];
    if (A2)
      for (int k = 0; k < Kd2; ++k)
        s += A2[(size_t)orow*lda2 + k] * B2[(size_t)k*ldb2 + bcol];
    accf[r] = s;
  }
#endif
  float bs = 0.f;
  if (bias)  bs += bias[bcol];
  if (bias2) bs += bias2[bcol];
  #pragma unroll
  for (int r = 0; r < 8; ++r){
    int orow = blockIdx.y*16 + r + 8*half;
    float v = accf[r] + bs;
    if (act == 1) v = v * sigm(1.702f*v);        // x*sigmoid(1.702x)
    C[(size_t)orow*ldc + bcol] = v;
  }
}

// ---------------------------------------------------------------------------
// Attention: one block per batch row b.
// sim_n = sum_k tanh(keys[b,n,k] + att_h[b,k]) * alphaW[k] + alphaB
// w = softmax(sim)*mask renormalized;  att = sum_n w_n * vals[b,n,:]
// ---------------------------------------------------------------------------
__global__ __launch_bounds__(256) void k_attn(const float* __restrict__ keys,
    const float* __restrict__ vals, const float* __restrict__ att_h,
    const float* __restrict__ alphaW, const float* __restrict__ alphaB,
    const float* __restrict__ masks, float* __restrict__ att){
  const int b = blockIdx.x;
  __shared__ float shw[NATT];
  const int tid = threadIdx.x, wave = tid>>5, lane = tid&31;
  const float* ah = att_h + (size_t)b*KK;
  for (int n = wave; n < NATT; n += 8){
    const float* kp = keys + ((size_t)b*NATT + n)*KK;
    float s = 0.f;
    for (int k = lane; k < KK; k += 32)
      s += fast_tanh(kp[k] + ah[k]) * alphaW[k];
    s += __shfl_xor(s,16,32); s += __shfl_xor(s,8,32); s += __shfl_xor(s,4,32);
    s += __shfl_xor(s,2,32);  s += __shfl_xor(s,1,32);
    if (lane == 0) shw[n] = s + alphaB[0];
  }
  __syncthreads();
  if (tid == 0){
    float mx = -1e30f;
    for (int n = 0; n < NATT; ++n) mx = fmaxf(mx, shw[n]);
    float den = 0.f;
    for (int n = 0; n < NATT; ++n) den += __expf(shw[n]-mx)*masks[b*NATT+n];
    float inv = 1.0f/den;
    for (int n = 0; n < NATT; ++n) shw[n] = __expf(shw[n]-mx)*masks[b*NATT+n]*inv;
  }
  __syncthreads();
  for (int j = tid; j < VDIM; j += 256){
    float a = 0.f;
    for (int n = 0; n < NATT; ++n)
      a += shw[n]*vals[((size_t)b*NATT + n)*VDIM + j];
    att[(size_t)b*VDIM + j] = a;
  }
}

// ---------------------------------------------------------------------------
// LSTM pointwise; also stores h_new as bf16 row t of hs (A-matrix of the final
// word-projection GEMM).
// ---------------------------------------------------------------------------
__global__ __launch_bounds__(256) void k_lstm(const float* __restrict__ s_all,
    const float* __restrict__ boom, const float* __restrict__ c_prev,
    float* __restrict__ h_cur, float* __restrict__ c_cur,
    unsigned short* __restrict__ hsb, int t){
  int idx = blockIdx.x*256 + threadIdx.x;   // 0..BB*HH-1
  int b = idx >> 9, j = idx & 511;
  const float* s = s_all + (size_t)b*4*HH;
  const float* g = boom  + (size_t)b*BOOMD;
  float ig = sigm(s[j]);
  float fg = sigm(s[HH + j]);
  float og = sigm(s[2*HH + j]);
  float attf = g[j] + g[HH+j] + g[2*HH+j] + g[3*HH+j];   // fold BOOM/H=4
  float it = s[3*HH + j] + attf;
  float c  = fg*c_prev[(size_t)b*HH + j] + ig*it;
  float h  = og*fast_tanh(c);
  c_cur[(size_t)b*HH + j] = c;
  h_cur[(size_t)b*HH + j] = h;
  hsb[(size_t)(t*BB + b)*HH + j] = f2bf(h);
}

// ---------------------------------------------------------------------------
// Fused word projection + partial softmax, bf16 WMMA (v_wmma_f32_16x16x32_bf16).
// Each WG owns a 64-column stripe of word_W: stage it bf16-packed in 64KB LDS,
// sweep all 62 row-tiles of hs (992x512 bf16, L2-hot).  A-fragment = two
// global_load_b128 per K-chunk (the 8 packed K-pairs a lane needs are two
// contiguous 16B spans).  Per row we emit partial[row][wg] = sum exp(logit)
// over the stripe (deterministic, no atomics) plus the biased label logit
// when the label column lives in this stripe.
// ---------------------------------------------------------------------------
__global__ __launch_bounds__(512) void k_word(const unsigned short* __restrict__ hsb,
    const float* __restrict__ wordW, const float* __restrict__ wordB,
    const int* __restrict__ labels, float* __restrict__ partial,
    float* __restrict__ labLogit){
  __shared__ unsigned ldsB[(HH/2)*NSTRIPE];   // pairs (k,k+1) x 64 cols = 64KB
  const int wg = blockIdx.x;
  const int n0 = wg * NSTRIPE;
  const int tid = threadIdx.x;
  for (int idx = tid; idx < (HH/2)*NSTRIPE; idx += 512){
    int n  = idx & (NSTRIPE-1);
    int kp = idx >> 6;
    int k  = kp*2;
    unsigned lo = f2bf(wordW[(size_t)k*NWORD + n0 + n]);
    unsigned hi = f2bf(wordW[(size_t)(k+1)*NWORD + n0 + n]);
    ldsB[idx] = lo | (hi << 16);
  }
  __syncthreads();
  const int wave = tid >> 5;
  const int lane = tid & 31;
  const int m    = lane & 15;
  const int half = lane >> 4;
  for (int mt = wave; mt < NROWS/16; mt += 16){
    v8f acc[4] = {};
    const int arow = mt*16 + m;
    const uvec4* ha = (const uvec4*)(hsb + (size_t)arow*HH);  // 8 bf16 per uvec4
    for (int kc = 0; kc < HH/32; ++kc){
      const int kb = kc*32;
      union { uvec4 q[2]; v16bf v; } ua;
      ua.q[0] = ha[kc*4 + 2*half];      // K = kb+16h .. kb+16h+7
      ua.q[1] = ha[kc*4 + 2*half + 1]; // K = kb+16h+8 .. +15
      const int khp = (kb + half*16) >> 1;   // K pair-index base for this half
      #pragma unroll
      for (int s = 0; s < 4; ++s){
        union { unsigned u[8]; v16bf v; } ub;
        #pragma unroll
        for (int v = 0; v < 8; ++v)
          ub.u[v] = ldsB[(khp + v)*NSTRIPE + s*16 + m];
        acc[s] = __builtin_amdgcn_wmma_f32_16x16x32_bf16(false, ua.v, false, ub.v,
                                                         (short)0, acc[s], false, false);
      }
    }
    // epilogue: bias, exp, 16-lane row reduction, label capture
    int lab[8];
    #pragma unroll
    for (int r = 0; r < 8; ++r) lab[r] = labels[mt*16 + r + 8*half];
    float rs[8];
    #pragma unroll
    for (int r = 0; r < 8; ++r) rs[r] = 0.f;
    #pragma unroll
    for (int s = 0; s < 4; ++s){
      const int col = n0 + s*16 + m;
      const float wb = wordB[col];
      #pragma unroll
      for (int r = 0; r < 8; ++r){
        float val = acc[s][r] + wb;
        if (lab[r] == col) labLogit[mt*16 + r + 8*half] = val;
        float e = __expf(val);
        e += __shfl_xor(e, 8, 16);
        e += __shfl_xor(e, 4, 16);
        e += __shfl_xor(e, 2, 16);
        e += __shfl_xor(e, 1, 16);
        rs[r] += e;
      }
    }
    if (m == 0){
      #pragma unroll
      for (int r = 0; r < 8; ++r)
        partial[(size_t)(mt*16 + r + 8*half)*NWG + wg] = rs[r];
    }
  }
}

// ---------------------------------------------------------------------------
// Deterministic fixed-order reduction of stripe partials; emit log-prob + mask.
// out[0:992]      = log_prob (B,31) row-major
// out[992:1984]   = mask     (B,31) row-major
// ---------------------------------------------------------------------------
__global__ __launch_bounds__(256) void k_final(const float* __restrict__ partial,
    const float* __restrict__ labLogit, const int* __restrict__ lens,
    float* __restrict__ out){
  int row = blockIdx.x*256 + threadIdx.x;
  if (row >= NROWS) return;
  float s = 0.f;
  for (int w = 0; w < NWG; ++w) s += partial[(size_t)row*NWG + w];
  int t = row >> 5, b = row & 31;
  out[b*NSTEP + t]         = labLogit[row] - __logf(s);
  out[NROWS + b*NSTEP + t] = ((t+1) < lens[b]) ? 1.0f : 0.0f;
}

// ---------------------------------------------------------------------------
extern "C" void kernel_launch(void* const* d_in, const int* in_sizes, int n_in,
                              void* d_out, int out_size, void* d_ws, size_t ws_size,
                              hipStream_t stream){
  (void)in_sizes; (void)n_in; (void)out_size; (void)ws_size;
  const float* init_h  = (const float*)d_in[0];
  const float* init_c  = (const float*)d_in[1];
  const float* att_fts = (const float*)d_in[2];
  const float* masks   = (const float*)d_in[3];
  const int*   y       = (const int*)  d_in[4];
  const int*   lens    = (const int*)  d_in[5];
  const float* embW    = (const float*)d_in[6];
  const float* Wk      = (const float*)d_in[7];
  const float* Wv      = (const float*)d_in[8];
  const float* h2kW    = (const float*)d_in[9];
  const float* h2kB    = (const float*)d_in[10];
  const float* alphaW  = (const float*)d_in[11];
  const float* alphaB  = (const float*)d_in[12];
  const float* a2bW    = (const float*)d_in[13];
  const float* a2bB    = (const float*)d_in[14];
  const float* i2hW    = (const float*)d_in[15];
  const float* i2hB    = (const float*)d_in[16];
  const float* h2hW    = (const float*)d_in[17];
  const float* h2hB    = (const float*)d_in[18];
  const float* wordW   = (const float*)d_in[19];
  const float* wordB   = (const float*)d_in[20];
  float* out = (float*)d_out;

  char* wp = (char*)d_ws;
  auto alloc = [&](size_t bytes)->void*{
    void* p = (void*)wp; wp += (bytes + 255) & ~(size_t)255; return p; };
  float* keys    = (float*)alloc((size_t)BB*NATT*KK*4);
  float* vals    = (float*)alloc((size_t)BB*NATT*VDIM*4);
  float* x_all   = (float*)alloc((size_t)NROWS*EE*4);
  int*   labels  = (int*)  alloc((size_t)NROWS*4);
  float* h_cur   = (float*)alloc((size_t)BB*HH*4);
  float* c_cur   = (float*)alloc((size_t)BB*HH*4);
  float* att_h   = (float*)alloc((size_t)BB*KK*4);
  float* attv    = (float*)alloc((size_t)BB*VDIM*4);
  float* s_all   = (float*)alloc((size_t)BB*4*HH*4);
  float* boom    = (float*)alloc((size_t)BB*BOOMD*4);
  unsigned short* hsb = (unsigned short*)alloc((size_t)NROWS*HH*2);
  float* partial = (float*)alloc((size_t)NROWS*NWG*4);
  float* labLog  = (float*)alloc((size_t)NROWS*4);

  // Phase 0: embeddings + labels
  k_gather<<<NROWS, 256, 0, stream>>>(y, embW, x_all, labels);

  // Phase 1: keys/vals precompute  (2048x1024)@(1024x512) x2, f32 WMMA
  k_gemm<<<dim3(KK/16, (BB*NATT)/16), 32, 0, stream>>>(att_fts, DATT, Wk, KK,
        nullptr, 0, nullptr, 0, nullptr, nullptr, keys, KK, DATT, 0, 0);
  k_gemm<<<dim3(VDIM/16, (BB*NATT)/16), 32, 0, stream>>>(att_fts, DATT, Wv, VDIM,
        nullptr, 0, nullptr, 0, nullptr, nullptr, vals, VDIM, DATT, 0, 0);

  // Phase 2: sequential recurrence (word projection hoisted out of the scan)
  for (int t = 0; t < NSTEP; ++t){
    const float* hp = (t == 0) ? init_h : h_cur;
    const float* cp = (t == 0) ? init_c : c_cur;
    // att_h = h @ h2key_W + b         (32x512x512)
    k_gemm<<<dim3(KK/16, BB/16), 32, 0, stream>>>(hp, HH, h2kW, KK,
          nullptr, 0, nullptr, 0, h2kB, nullptr, att_h, KK, HH, 0, 0);
    // attention scores + context
    k_attn<<<BB, 256, 0, stream>>>(keys, vals, att_h, alphaW, alphaB, masks, attv);
    // s_all = x@i2h + h@h2h + biases  (32x512x2048, dual product)
    k_gemm<<<dim3((4*HH)/16, BB/16), 32, 0, stream>>>(x_all + (size_t)t*BB*EE, EE,
          i2hW, 4*HH, hp, HH, h2hW, 4*HH, i2hB, h2hB, s_all, 4*HH, EE, HH, 0);
    // boom = silu-gate(att@a2b + b)   (32x512x2048)
    k_gemm<<<dim3(BOOMD/16, BB/16), 32, 0, stream>>>(attv, VDIM, a2bW, BOOMD,
          nullptr, 0, nullptr, 0, a2bB, nullptr, boom, BOOMD, VDIM, 0, 1);
    // LSTM pointwise; stash h_new (f32 + bf16)
    k_lstm<<<(BB*HH)/256, 256, 0, stream>>>(s_all, boom, cp, h_cur, c_cur, hsb, t);
  }

  // Phase 3: batched word projection (992x512x32000, bf16 WMMA) + log_softmax
  k_word<<<NWG, 512, 0, stream>>>(hsb, wordW, wordB, labels, partial, labLog);
  k_final<<<(NROWS + 255)/256, 256, 0, stream>>>(partial, labLog, lens, out);
}
